// Update_8529805050031
// MI455X (gfx1250) — compile-verified, matrix operand order checked by reading, not run on hardware
//
#include <hip/hip_runtime.h>
#include <hip/hip_bf16.h>

typedef __attribute__((ext_vector_type(16))) _Float16 v16h;
typedef __attribute__((ext_vector_type(8)))  _Float16 v8h;
typedef __attribute__((ext_vector_type(8)))  float    v8f;

#define DIMX    384
#define NFRAMES 64
#define NPATCH  2048
#define NEDGE   (NPATCH * NFRAMES)   // 131072
#define LN_EPS  1e-3f

#define BM 128
#define BN 128
#define BK 32

// ---------------------------------------------------------------------------
// wave32 reduction helper
// ---------------------------------------------------------------------------
__device__ __forceinline__ float wave_sum(float v) {
#pragma unroll
    for (int o = 16; o > 0; o >>= 1) v += __shfl_xor(v, o, 32);
    return v;
}

// ---------------------------------------------------------------------------
// f32 -> f16 weight conversion (weights stay L2-resident; done once per launch)
// ---------------------------------------------------------------------------
__global__ void cvt_f16_kernel(const float* __restrict__ src,
                               _Float16* __restrict__ dst, int n) {
    int i = blockIdx.x * 256 + threadIdx.x;
    if (i < n) dst[i] = (_Float16)src[i];
}

// ---------------------------------------------------------------------------
// WMMA GEMM:  out[M,N] = epilogue( X[M,K] @ W[N,K]^T + bias )
//   epilogue: act (0 none / 1 relu / 2 sigmoid), then *mul (gate), then +res.
// Block: 256 thr = 8 waves, tile 128(M) x 128(N), K double-buffered 32-slabs.
// Wave (wm 0..3, wn 0..1): 32x64 strip -> 2 A-frags x 4 B-frags = 8 WMMA/slab.
// W slab staged with GLOBAL_LOAD_ASYNC_TO_LDS_B128 (ASYNCcnt) when 16B-aligned.
// ---------------------------------------------------------------------------
__global__ __launch_bounds__(256) void gemm_wmma_kernel(
    const float* __restrict__ X, const _Float16* __restrict__ W,
    const float* __restrict__ bias, const float* __restrict__ res,
    const float* __restrict__ mul, float* __restrict__ out,
    int M, int N, int K, int act)
{
    __shared__ __align__(16) _Float16 As[2][BM][BK];   // 2 x 8 KB
    __shared__ __align__(16) _Float16 Ws[2][BN][BK];   // 2 x 8 KB

    const int tid  = threadIdx.x;
    const int lane = tid & 31;
    const int wave = tid >> 5;
    const int wm   = (wave & 3) << 5;   // 0,32,64,96
    const int wn   = (wave >> 2) << 6;  // 0,64
    const int m0   = blockIdx.x * BM;
    const int n0   = blockIdx.y * BN;

    v8f acc[2][4] = {};

    // WMMA 16-bit A/B fragment addressing (16x32):
    //   lane<16: K 0..7,16..23 ; lane>=16: K 8..15,24..31 ; row/col = lane&15
    const int arow = wm + (lane & 15);
    const int kb   = (lane >> 4) << 3;           // 0 or 8 (halfs)

    // staging mapping: 128 rows x 32 cols, 256 threads -> 16 elems each
    const int sr = tid >> 1;            // 0..127
    const int sc = (tid & 1) << 4;      // 0 or 16
    const bool wAsync = ((K & 7) == 0); // 16B-aligned f16 rows (K=384 layers)

    auto stageA = [&](int buf, int k0) {
        const float* xp = X + (size_t)(m0 + sr) * K + k0 + sc;
        if (k0 + BK <= K) {
            if ((K & 3) == 0) {           // 16B-aligned f32 rows
#pragma unroll
                for (int j = 0; j < 4; ++j) {
                    float4 t = *(const float4*)(xp + 4 * j);
                    As[buf][sr][sc + 4*j + 0] = (_Float16)t.x;
                    As[buf][sr][sc + 4*j + 1] = (_Float16)t.y;
                    As[buf][sr][sc + 4*j + 2] = (_Float16)t.z;
                    As[buf][sr][sc + 4*j + 3] = (_Float16)t.w;
                }
            } else {                      // K even -> 8B-aligned
#pragma unroll
                for (int j = 0; j < 8; ++j) {
                    float2 t = *(const float2*)(xp + 2 * j);
                    As[buf][sr][sc + 2*j + 0] = (_Float16)t.x;
                    As[buf][sr][sc + 2*j + 1] = (_Float16)t.y;
                }
            }
        } else {                          // single ragged tail slab (corr1)
#pragma unroll
            for (int j = 0; j < 16; ++j) {
                int kx = k0 + sc + j;
                As[buf][sr][sc + j] = (kx < K) ? (_Float16)xp[j] : (_Float16)0.f;
            }
        }
    };

    auto stageW = [&](int buf, int k0) {
        const _Float16* wp = W + (size_t)(n0 + sr) * K + k0 + sc;
        if (wAsync && (k0 + BK <= K)) {
            // pure f16 global->LDS copy: CDNA5 async DMA (2 x 16B per thread)
            unsigned lds = (unsigned)(unsigned long long)&Ws[buf][sr][sc];
            unsigned long long ga = (unsigned long long)wp;
            asm volatile(
                "global_load_async_to_lds_b128 %0, %1, off\n\t"
                "global_load_async_to_lds_b128 %0, %1, off offset:16"
                :: "v"(lds), "v"(ga) : "memory");
        } else if (k0 + BK <= K) {        // K even -> 4B-aligned dword copies
            const unsigned* s32 = (const unsigned*)wp;
            unsigned* d32 = (unsigned*)&Ws[buf][sr][sc];
#pragma unroll
            for (int j = 0; j < 8; ++j) d32[j] = s32[j];
        } else {
#pragma unroll
            for (int j = 0; j < 16; ++j) {
                int kx = k0 + sc + j;
                Ws[buf][sr][sc + j] = (kx < K) ? wp[j] : (_Float16)0.f;
            }
        }
    };

    const int nIter = (K + BK - 1) / BK;
    stageA(0, 0);
    stageW(0, 0);
    asm volatile("s_wait_asynccnt 0x0" ::: "memory");
    __syncthreads();

    for (int it = 0; it < nIter; ++it) {
        const int cur = it & 1;
        if (it + 1 < nIter) {             // overlap next-slab staging with WMMA
            stageA(cur ^ 1, (it + 1) * BK);
            stageW(cur ^ 1, (it + 1) * BK);
        }
        // gather all fragments first, then issue WMMAs back-to-back
        v16h a[2], b[4];
#pragma unroll
        for (int i = 0; i < 2; ++i) {
            v8h a0 = *(const v8h*)&As[cur][arow + 16 * i][kb];
            v8h a1 = *(const v8h*)&As[cur][arow + 16 * i][kb + 16];
            a[i] = __builtin_shufflevector(a0, a1, 0,1,2,3,4,5,6,7,
                                                   8,9,10,11,12,13,14,15);
        }
#pragma unroll
        for (int t = 0; t < 4; ++t) {
            int nr = wn + t * 16 + (lane & 15);
            v8h b0 = *(const v8h*)&Ws[cur][nr][kb];
            v8h b1 = *(const v8h*)&Ws[cur][nr][kb + 16];
            b[t] = __builtin_shufflevector(b0, b1, 0,1,2,3,4,5,6,7,
                                                   8,9,10,11,12,13,14,15);
        }
#pragma unroll
        for (int i = 0; i < 2; ++i)
#pragma unroll
            for (int t = 0; t < 4; ++t)
                acc[i][t] = __builtin_amdgcn_wmma_f32_16x16x32_f16(
                    false, a[i], false, b[t], (short)0, acc[i][t], false, false);

        asm volatile("s_wait_asynccnt 0x0" ::: "memory");
        __syncthreads();
    }

    // ---- fused epilogue (C/D layout: VGPR r <-> M = r + 8*(lane>=16)) ----
    const int rb = m0 + wm + ((lane >> 4) << 3);
#pragma unroll
    for (int i = 0; i < 2; ++i) {
#pragma unroll
        for (int t = 0; t < 4; ++t) {
            const int col = n0 + wn + t * 16 + (lane & 15);
            const float bv = bias ? bias[col] : 0.f;
#pragma unroll
            for (int r = 0; r < 8; ++r) {
                size_t o = (size_t)(rb + 16 * i + r) * N + col;
                float v = acc[i][t][r] + bv;
                if (act == 1)      v = fmaxf(v, 0.f);
                else if (act == 2) v = 1.f / (1.f + __expf(-v));
                if (mul) v *= mul[o];
                if (res) v += res[o];
                out[o] = v;
            }
        }
    }
}

// ---------------------------------------------------------------------------
// Row LayerNorm (384 cols, 1 wave/row, 8 rows/block). Optional 2 extra addends
// and trailing ReLU. In-place safe (row fully register-resident).
// ---------------------------------------------------------------------------
__global__ __launch_bounds__(256) void ln_kernel(
    const float* __restrict__ in, const float* __restrict__ add1,
    const float* __restrict__ add2, const float* __restrict__ gamma,
    const float* __restrict__ beta, float* __restrict__ out, int relu)
{
    const int row  = blockIdx.x * 8 + (threadIdx.x >> 5);
    const int lane = threadIdx.x & 31;
    const size_t base = (size_t)row * DIMX;
    float x[12];
#pragma unroll
    for (int j = 0; j < 12; ++j) {
        int c = lane + j * 32;
        float v = in[base + c];
        if (add1) v += add1[base + c];
        if (add2) v += add2[base + c];
        x[j] = v;
    }
    float s = 0.f;
#pragma unroll
    for (int j = 0; j < 12; ++j) s += x[j];
    const float mean = wave_sum(s) * (1.f / DIMX);
    float vs = 0.f;
#pragma unroll
    for (int j = 0; j < 12; ++j) { float d = x[j] - mean; vs += d * d; }
    const float inv = rsqrtf(wave_sum(vs) * (1.f / DIMX) + LN_EPS);
#pragma unroll
    for (int j = 0; j < 12; ++j) {
        int c = lane + j * 32;
        float nv = (x[j] - mean) * inv * gamma[c] + beta[c];
        if (relu) nv = fmaxf(nv, 0.f);
        out[base + c] = nv;
    }
}

// ---------------------------------------------------------------------------
// neighbor gather with mask: out[e] = (idx[e]>=0) ? net[idx[e]] : 0
// ---------------------------------------------------------------------------
__global__ void gather_mask_kernel(const float* __restrict__ net,
                                   const int* __restrict__ idx,
                                   float* __restrict__ out) {
    const int e = blockIdx.x, c = threadIdx.x;
    const int i = idx[e];
    out[(size_t)e * DIMX + c] = (i >= 0) ? net[(size_t)i * DIMX + c] : 0.f;
}

// ---------------------------------------------------------------------------
// SoftAgg segment reductions (segments are structurally regular):
//   kk: segment s = 64 consecutive edges [s*64, s*64+64)
//   ij: segment s = ii*64+jj = 32 edges { ii*2048 + p*64 + jj }
// y[s] = sum_r f[r] * softmax_r(g[r])     (one block of 384 thr per segment)
// ---------------------------------------------------------------------------
__global__ void segreduce_kk_kernel(const float* __restrict__ f,
                                    const float* __restrict__ g,
                                    float* __restrict__ y) {
    const int s = blockIdx.x, col = threadIdx.x;
    const size_t base = (size_t)s * NFRAMES * DIMX + col;
    float mx = -3.402823466e38f;
    for (int r = 0; r < NFRAMES; ++r)
        mx = fmaxf(mx, g[base + (size_t)r * DIMX]);
    float den = 0.f, acc = 0.f;
    for (int r = 0; r < NFRAMES; ++r) {
        size_t o = base + (size_t)r * DIMX;
        float w = __expf(g[o] - mx);
        den += w; acc += f[o] * w;
    }
    y[(size_t)s * DIMX + col] = acc / den;
}

__global__ void segreduce_ij_kernel(const float* __restrict__ f,
                                    const float* __restrict__ g,
                                    float* __restrict__ y) {
    const int s = blockIdx.x, col = threadIdx.x;
    const int fi = s >> 6, fj = s & 63;
    const size_t row0 = (size_t)fi * 2048 + fj;
    float mx = -3.402823466e38f;
    for (int p = 0; p < 32; ++p)
        mx = fmaxf(mx, g[(row0 + (size_t)p * 64) * DIMX + col]);
    float den = 0.f, acc = 0.f;
    for (int p = 0; p < 32; ++p) {
        size_t o = (row0 + (size_t)p * 64) * DIMX + col;
        float w = __expf(g[o] - mx);
        den += w; acc += f[o] * w;
    }
    y[(size_t)s * DIMX + col] = acc / den;
}

__global__ void expand_add_kk_kernel(float* __restrict__ net,
                                     const float* __restrict__ yh,
                                     const int* __restrict__ kk) {
    const int e = blockIdx.x, c = threadIdx.x;
    net[(size_t)e * DIMX + c] += yh[(size_t)kk[e] * DIMX + c];
}

__global__ void expand_add_ij_kernel(float* __restrict__ net,
                                     const float* __restrict__ yh,
                                     const int* __restrict__ ii,
                                     const int* __restrict__ jj) {
    const int e = blockIdx.x, c = threadIdx.x;
    const int s = ii[e] * NFRAMES + jj[e];
    net[(size_t)e * DIMX + c] += yh[(size_t)s * DIMX + c];
}

// ---------------------------------------------------------------------------
// heads: r = relu(net); d = r@dw^T+db ; w = sigmoid(r@ww^T+wb). 1 wave/edge.
// ---------------------------------------------------------------------------
__global__ __launch_bounds__(256) void heads_kernel(
    const float* __restrict__ net,
    const float* __restrict__ dw, const float* __restrict__ db,
    const float* __restrict__ ww, const float* __restrict__ wb,
    float* __restrict__ dOut, float* __restrict__ wOut)
{
    const int e = blockIdx.x * 8 + (threadIdx.x >> 5);
    const int lane = threadIdx.x & 31;
    const size_t base = (size_t)e * DIMX;
    float s0 = 0.f, s1 = 0.f, s2 = 0.f, s3 = 0.f;
#pragma unroll
    for (int j = 0; j < 12; ++j) {
        int c = lane + j * 32;
        float r = fmaxf(net[base + c], 0.f);
        s0 += r * dw[c];       s1 += r * dw[DIMX + c];
        s2 += r * ww[c];       s3 += r * ww[DIMX + c];
    }
    s0 = wave_sum(s0); s1 = wave_sum(s1);
    s2 = wave_sum(s2); s3 = wave_sum(s3);
    if (lane == 0) {
        dOut[(size_t)e * 2 + 0] = s0 + db[0];
        dOut[(size_t)e * 2 + 1] = s1 + db[1];
        wOut[(size_t)e * 2 + 0] = 1.f / (1.f + __expf(-(s2 + wb[0])));
        wOut[(size_t)e * 2 + 1] = 1.f / (1.f + __expf(-(s3 + wb[1])));
    }
}

// ---------------------------------------------------------------------------
// host orchestration
// ---------------------------------------------------------------------------
extern "C" void kernel_launch(void* const* d_in, const int* in_sizes, int n_in,
                              void* d_out, int out_size, void* d_ws, size_t ws_size,
                              hipStream_t stream) {
    (void)in_sizes; (void)n_in; (void)out_size; (void)ws_size;
    auto F = [&](int i) { return (const float*)d_in[i]; };
    auto I = [&](int i) { return (const int*)d_in[i]; };

    const float* net_in = F(0);
    const float* inp    = F(1);
    const float* corr   = F(2);
    const int*   ii     = I(4);
    const int*   jj     = I(5);
    const int*   kk     = I(6);
    const int*   ix     = I(7);
    const int*   jx     = I(8);

    // params (dict insertion order), d_in index:
    // 9 corr1_w 10 corr1_b 11 corr2_w 12 corr2_b 13 corr_ln_g 14 corr_ln_b
    // 15 corr3_w 16 corr3_b 17 norm_g 18 norm_b 19..26 c1/c2 27..38 agg
    // 39..42 gru_ln 43..54 gr1/gr2 55 d_w 56 d_b 57 w_w 58 w_b
    enum { W_CORR1, W_CORR2, W_CORR3, W_C1_1, W_C1_2, W_C2_1, W_C2_2,
           W_KK_F, W_KK_G, W_KK_H, W_IJ_F, W_IJ_G, W_IJ_H,
           W_GR1_GATE, W_GR1_R1, W_GR1_R2, W_GR2_GATE, W_GR2_R1, W_GR2_R2,
           NW };
    static const int widx[NW] = {9,11,15,19,21,23,25,27,29,31,33,35,37,
                                 43,45,47,49,51,53};
    const int K0 = 2 * 49 * 3 * 3;  // 882
    int wsz[NW]; wsz[0] = DIMX * K0;
    for (int i = 1; i < NW; ++i) wsz[i] = DIMX * DIMX;

    // ---- workspace layout ----
    char* ws = (char*)d_ws;
    _Float16* wh[NW];
    size_t off = 0;
    for (int i = 0; i < NW; ++i) { wh[i] = (_Float16*)(ws + off); off += (size_t)wsz[i] * 2; }
    off = (off + 255) & ~(size_t)255;
    float* buf1 = (float*)(ws + off); off += (size_t)NEDGE * DIMX * 4;
    float* buf2 = (float*)(ws + off); off += (size_t)NEDGE * DIMX * 4;
    float* y1   = (float*)(ws + off); off += (size_t)NPATCH * DIMX * 4;
    float* yh1  = (float*)(ws + off); off += (size_t)NPATCH * DIMX * 4;
    float* y2   = (float*)(ws + off); off += (size_t)NFRAMES * NFRAMES * DIMX * 4;
    float* yh2  = (float*)(ws + off);

    float* outNet = (float*)d_out;
    float* outD   = outNet + (size_t)NEDGE * DIMX;
    float* outW   = outD + (size_t)NEDGE * 2;

    // ---- weight f32 -> f16 (once; tiny) ----
    for (int i = 0; i < NW; ++i)
        cvt_f16_kernel<<<(wsz[i] + 255) / 256, 256, 0, stream>>>(F(widx[i]), wh[i], wsz[i]);

    auto gemm = [&](const float* X, const _Float16* Wh, const float* bias,
                    const float* res, const float* mul, float* out,
                    int M, int N, int K, int act) {
        dim3 grid(M / BM, N / BN);
        gemm_wmma_kernel<<<grid, 256, 0, stream>>>(X, Wh, bias, res, mul, out,
                                                   M, N, K, act);
    };
    const int E = NEDGE;
    const int lnGrid = E / 8;

    // ---- corr encoder: Lin+ReLU, Lin, LN+ReLU, Lin ----
    gemm(corr, wh[W_CORR1], F(10), nullptr, nullptr, buf1, E, DIMX, K0, 1);
    gemm(buf1, wh[W_CORR2], F(12), nullptr, nullptr, buf2, E, DIMX, DIMX, 0);
    ln_kernel<<<lnGrid, 256, 0, stream>>>(buf2, nullptr, nullptr, F(13), F(14), buf1, 1);
    gemm(buf1, wh[W_CORR3], F(16), nullptr, nullptr, buf2, E, DIMX, DIMX, 0);
    // net = LN(net_in + inp + c)
    ln_kernel<<<lnGrid, 256, 0, stream>>>(net_in, inp, buf2, F(17), F(18), outNet, 0);

    // ---- neighbor MLP residuals (c1 with ix, c2 with jx) ----
    gather_mask_kernel<<<E, DIMX, 0, stream>>>(outNet, ix, buf1);
    gemm(buf1, wh[W_C1_1], F(20), nullptr, nullptr, buf2, E, DIMX, DIMX, 1);
    gemm(buf2, wh[W_C1_2], F(22), outNet, nullptr, outNet, E, DIMX, DIMX, 0);
    gather_mask_kernel<<<E, DIMX, 0, stream>>>(outNet, jx, buf1);
    gemm(buf1, wh[W_C2_1], F(24), nullptr, nullptr, buf2, E, DIMX, DIMX, 1);
    gemm(buf2, wh[W_C2_2], F(26), outNet, nullptr, outNet, E, DIMX, DIMX, 0);

    // ---- SoftAgg over kk ----
    gemm(outNet, wh[W_KK_F], F(28), nullptr, nullptr, buf1, E, DIMX, DIMX, 0);
    gemm(outNet, wh[W_KK_G], F(30), nullptr, nullptr, buf2, E, DIMX, DIMX, 0);
    segreduce_kk_kernel<<<NPATCH, DIMX, 0, stream>>>(buf1, buf2, y1);
    gemm(y1, wh[W_KK_H], F(32), nullptr, nullptr, yh1, NPATCH, DIMX, DIMX, 0);
    expand_add_kk_kernel<<<E, DIMX, 0, stream>>>(outNet, yh1, kk);

    // ---- SoftAgg over (ii,jj) ----
    gemm(outNet, wh[W_IJ_F], F(34), nullptr, nullptr, buf1, E, DIMX, DIMX, 0);
    gemm(outNet, wh[W_IJ_G], F(36), nullptr, nullptr, buf2, E, DIMX, DIMX, 0);
    segreduce_ij_kernel<<<NFRAMES * NFRAMES, DIMX, 0, stream>>>(buf1, buf2, y2);
    gemm(y2, wh[W_IJ_H], F(38), nullptr, nullptr, yh2, NFRAMES * NFRAMES, DIMX, DIMX, 0);
    expand_add_ij_kernel<<<E, DIMX, 0, stream>>>(outNet, yh2, ii, jj);

    // ---- GRU: LN, GatedResidual, LN, GatedResidual ----
    ln_kernel<<<lnGrid, 256, 0, stream>>>(outNet, nullptr, nullptr, F(39), F(40), outNet, 0);
    gemm(outNet, wh[W_GR1_GATE], F(44), nullptr, nullptr, buf1, E, DIMX, DIMX, 2);
    gemm(outNet, wh[W_GR1_R1],   F(46), nullptr, nullptr, buf2, E, DIMX, DIMX, 1);
    gemm(buf2,   wh[W_GR1_R2],   F(48), outNet, buf1,     outNet, E, DIMX, DIMX, 0);
    ln_kernel<<<lnGrid, 256, 0, stream>>>(outNet, nullptr, nullptr, F(41), F(42), outNet, 0);
    gemm(outNet, wh[W_GR2_GATE], F(50), nullptr, nullptr, buf1, E, DIMX, DIMX, 2);
    gemm(outNet, wh[W_GR2_R1],   F(52), nullptr, nullptr, buf2, E, DIMX, DIMX, 1);
    gemm(buf2,   wh[W_GR2_R2],   F(54), outNet, buf1,     outNet, E, DIMX, DIMX, 0);

    // ---- heads: d (grad_clip = identity fwd), w = sigmoid ----
    heads_kernel<<<E / 8, 256, 0, stream>>>(outNet, F(55), F(56), F(57), F(58),
                                            outD, outW);
}
